// ChebyshevBasis_1606317769440
// MI455X (gfx1250) — matrix operand order repaired and checked by reading, not compiled, not run on hardware
//
#include <hip/hip_runtime.h>
#include <hip/hip_bf16.h>

// Problem constants (match reference)
#define B_ 8192
#define F_ 512
#define O_ 512
#define NB 16            // slots per feature: [x, T1..T15]; T0 folded into bias
#define K_ (F_ * NB)     // 8192

// GEMM tiling
#define BM 128
#define BN 128
#define KB 64            // 4 features * 16 slots per K-chunk
#define FB (KB / NB)     // 4 features per chunk
#define KC (K_ / KB)     // 128 K-iterations
#define LDA (KB + 8)     // LDS row stride in halves (144B, 16B-aligned, conflict padding)
#define LDB (KB + 8)

typedef __attribute__((ext_vector_type(16))) _Float16 v16h;
typedef __attribute__((ext_vector_type(8)))  float    v8f;

union V16H { v16h v; float4 f4[2]; };

// ---------------------------------------------------------------------------
// CDNA5 feature shims (guarded so compilation never breaks)
// ---------------------------------------------------------------------------
#if __has_builtin(__builtin_amdgcn_global_load_async_to_lds_b128)
#define HAVE_ASYNC_LDS 1
#endif

#if defined(HAVE_ASYNC_LDS)
typedef int v4i __attribute__((vector_size(16)));
typedef __attribute__((address_space(1))) v4i* as1_v4i_ptr;   // global
typedef __attribute__((address_space(3))) v4i* as3_v4i_ptr;   // LDS
#endif

__device__ __forceinline__ void async_copy16(const void* gsrc, void* ldst) {
#if defined(HAVE_ASYNC_LDS)
    // GLOBAL_LOAD_ASYNC_TO_LDS_B128: bypasses VGPRs, tracked by ASYNCcnt
    __builtin_amdgcn_global_load_async_to_lds_b128(
        (as1_v4i_ptr)gsrc, (as3_v4i_ptr)ldst, 0, 0);
#else
    *reinterpret_cast<float4*>(ldst) = *reinterpret_cast<const float4*>(gsrc);
#endif
}

__device__ __forceinline__ void wait_async0() {
#if defined(HAVE_ASYNC_LDS)
#if __has_builtin(__builtin_amdgcn_s_wait_asynccnt)
    __builtin_amdgcn_s_wait_asynccnt(0);
#else
    asm volatile("s_wait_asynccnt 0x0" ::: "memory");
#endif
#endif
}

__device__ __forceinline__ float fast_tanh(float v) {
#if __has_builtin(__builtin_amdgcn_tanhf)
    return __builtin_amdgcn_tanhf(v);   // v_tanh_f32 (gfx1250 TRANS op)
#else
    return tanhf(v);
#endif
}

// ---------------------------------------------------------------------------
// Kernel 1: pack W[o][k] (f16, o-major) from coeffs + base_weight.
//   k = i*16 + n ; n==0 -> base_weight[i,o] (residual slot), n>=1 -> coeffs[i,o,n]
// ---------------------------------------------------------------------------
__global__ __launch_bounds__(256) void pack_w_kernel(const float* __restrict__ coeffs,
                                                     const float* __restrict__ bw,
                                                     _Float16* __restrict__ Wp) {
    int idx = blockIdx.x * 256 + threadIdx.x;          // 0 .. O_*K_-1
    int o = idx >> 13;                                  // / K_
    int k = idx & (K_ - 1);
    int i = k >> 4;
    int n = k & 15;
    float v = (n == 0) ? bw[i * O_ + o] : coeffs[(size_t)(i * O_ + o) * NB + n];
    Wp[idx] = (_Float16)v;
}

// ---------------------------------------------------------------------------
// Kernel 2: bias[o] = sum_i coeffs[i,o,0]  (folded T0 term); also writes kl=0.
// ---------------------------------------------------------------------------
__global__ __launch_bounds__(256) void bias_kl_kernel(const float* __restrict__ coeffs,
                                                      float* __restrict__ bias,
                                                      float* __restrict__ out) {
    int o = blockIdx.x * 256 + threadIdx.x;
    if (o < O_) {
        float s = 0.f;
        for (int i = 0; i < F_; ++i) s += coeffs[(size_t)(i * O_ + o) * NB];
        bias[o] = s;
    }
    if (o == 0) out[(size_t)B_ * O_] = 0.0f;            // kl scalar
}

// ---------------------------------------------------------------------------
// Kernel 3: fused Chebyshev-basis + GEMM, double-buffered LDS pipeline.
//   out[b,o] = bias[o] + sum_k A[b,k] * W[o,k]
// ---------------------------------------------------------------------------
__global__ __launch_bounds__(256) void cheb_gemm_kernel(const float* __restrict__ x,
                                                        const _Float16* __restrict__ Wp,
                                                        const float* __restrict__ bias,
                                                        float* __restrict__ out) {
    __shared__ _Float16 Asm[2][BM * LDA];   // 2 x 18 KB
    __shared__ _Float16 Bsm[2][BN * LDB];   // 2 x 18 KB

    const int tid  = threadIdx.x;
    const int o0   = blockIdx.x * BN;
    const int b0   = blockIdx.y * BM;
    const int wave = tid >> 5;
    const int lane = tid & 31;
    const int wr   = wave & 3;           // row group: 32 rows each
    const int wc   = wave >> 2;          // col group: 64 cols each

    // wave32 WMMA fragment lane roles (per ISA 16-bit layouts)
    const int rowA  = lane & 15;
    const int koffA = (lane < 16) ? 0 : 8;    // A: halves[0..7]=K+off, halves[8..15]=K+off+16
    const int colB  = lane & 15;
    const int koffB = (lane < 16) ? 0 : 8;    // B: halves[0..7]=K+off, halves[8..15]=K+off+16

    v8f acc[2][4];
#pragma unroll
    for (int tm = 0; tm < 2; ++tm)
#pragma unroll
        for (int tn = 0; tn < 4; ++tn) acc[tm][tn] = (v8f){0.f,0.f,0.f,0.f,0.f,0.f,0.f,0.f};

    // ---- tile loader: W tile via async global->LDS, A tile computed on the fly
    auto load_tile = [&](int kc, int buf) {
        // W tile: 128 rows x 64 halves = 1024 x 16B chunks, async to LDS
#pragma unroll
        for (int j = 0; j < 4; ++j) {
            int p  = tid + j * 256;
            int r  = p >> 3;             // 0..127
            int c4 = p & 7;              // 0..7
            async_copy16(Wp + (size_t)(o0 + r) * K_ + kc * KB + c4 * 8,
                         &Bsm[buf][r * LDB + c4 * 8]);
        }
        // A tile: 128 rows x 4 features -> 16 f16 slots each (x, T1..T15)
        const int i0 = kc * FB;
#pragma unroll
        for (int j = 0; j < 2; ++j) {
            int p   = tid + j * 256;     // 0..511
            int row = p >> 2;            // 0..127
            int f   = p & 3;             // 0..3
            float xv = x[(size_t)(b0 + row) * F_ + i0 + f];
            float t  = fast_tanh(xv);
            union { _Float16 h[16]; float4 f4[2]; } u;
            u.h[0] = (_Float16)xv;       // residual slot
            u.h[1] = (_Float16)t;        // T1
            float Tm2 = 1.0f, Tm1 = t;
#pragma unroll
            for (int n = 2; n < 16; ++n) {
                float T = 2.0f * t * Tm1 - Tm2;
                u.h[n] = (_Float16)T;
                Tm2 = Tm1; Tm1 = T;
            }
            float4* dst = reinterpret_cast<float4*>(&Asm[buf][row * LDA + f * NB]);
            dst[0] = u.f4[0];
            dst[1] = u.f4[1];
        }
    };

    // ---- pipeline prologue
    load_tile(0, 0);
    wait_async0();
    __syncthreads();

    for (int kc = 0; kc < KC; ++kc) {
        const int cur = kc & 1;

        // issue next tile (async W loads fly during the WMMA block below)
        if (kc + 1 < KC) load_tile(kc + 1, cur ^ 1);

        // ---- MMA on current buffer: two 32-wide K-steps over the 64-half chunk
#pragma unroll
        for (int ks = 0; ks < 2; ++ks) {
            const int kb = ks * 32;
            V16H a[2], b[4];
#pragma unroll
            for (int tm = 0; tm < 2; ++tm) {
                const _Float16* ap = &Asm[cur][(wr * 32 + tm * 16 + rowA) * LDA + kb + koffA];
                a[tm].f4[0] = *reinterpret_cast<const float4*>(ap);
                a[tm].f4[1] = *reinterpret_cast<const float4*>(ap + 16);
            }
#pragma unroll
            for (int tn = 0; tn < 4; ++tn) {
                const _Float16* bp = &Bsm[cur][(wc * 64 + tn * 16 + colB) * LDB + kb + koffB];
                b[tn].f4[0] = *reinterpret_cast<const float4*>(bp);
                b[tn].f4[1] = *reinterpret_cast<const float4*>(bp + 16);
            }
#pragma unroll
            for (int tm = 0; tm < 2; ++tm)
#pragma unroll
                for (int tn = 0; tn < 4; ++tn)
                    acc[tm][tn] = __builtin_amdgcn_wmma_f32_16x16x32_f16(
                        false, a[tm].v, false, b[tn].v,
                        (short)0, acc[tm][tn], false, false);
        }

        // next buffer must be fully resident (async W + DS-stored A) before reuse
        wait_async0();
        __syncthreads();
    }

    // ---- epilogue: add bias, store f32
#pragma unroll
    for (int tn = 0; tn < 4; ++tn) {
        const int col = o0 + wc * 64 + tn * 16 + (lane & 15);
        const float bv = bias[col];
#pragma unroll
        for (int tm = 0; tm < 2; ++tm) {
            const int rbase = b0 + wr * 32 + tm * 16 + ((lane < 16) ? 0 : 8);
#pragma unroll
            for (int v = 0; v < 8; ++v)
                out[(size_t)(rbase + v) * O_ + col] = acc[tm][tn][v] + bv;
        }
    }
}

// ---------------------------------------------------------------------------
extern "C" void kernel_launch(void* const* d_in, const int* in_sizes, int n_in,
                              void* d_out, int out_size, void* d_ws, size_t ws_size,
                              hipStream_t stream) {
    const float* x      = (const float*)d_in[0];
    const float* coeffs = (const float*)d_in[1];
    const float* bw     = (const float*)d_in[2];
    float* out = (float*)d_out;

    _Float16* Wp  = (_Float16*)d_ws;                                   // 8 MB
    float*    bia = (float*)((char*)d_ws + (size_t)O_ * K_ * sizeof(_Float16));

    pack_w_kernel<<<(O_ * K_) / 256, 256, 0, stream>>>(coeffs, bw, Wp);
    bias_kl_kernel<<<(O_ + 255) / 256, 256, 0, stream>>>(coeffs, bia, out);

    dim3 grid(O_ / BN, B_ / BM);   // 4 x 64 blocks
    cheb_gemm_kernel<<<grid, 256, 0, stream>>>(x, Wp, bia, out);
}
